// DRGCNLayer_72258529788047
// MI455X (gfx1250) — matrix operand order repaired and checked by reading, not compiled, not run on hardware
//
#include <hip/hip_runtime.h>
#include <hip/hip_bf16.h>

// DRGCN temporal graph attention for gfx1250 (MI455X).
// - All dense GEMMs (edge MLP 257->128, K/V projections, node Q projection)
//   run on v_wmma_f32_16x16x32_bf16 (wave32, M = 16 edges per wave tile).
// - Weight B-panels are staged into LDS once per block via the Tensor Data
//   Mover (tensor_load_to_lds, TENSORcnt) so B-fragments come from ds_load
//   instead of redundant per-wave global reads.
// Workspace (~54.6 MB): Q fp32 [50000,128], scores fp32 [800000,8],
//   segmax/denom fp32 [50000,8], bf16 Wq/Wk/Wv [128,128], W1 [257,128].

#define NN 50000
#define NE 800000
#define HID 128
#define NH  8

typedef __attribute__((ext_vector_type(16))) __bf16 v16bf;
typedef __attribute__((ext_vector_type(8)))  float  v8f;

#if __has_builtin(__builtin_amdgcn_tensor_load_to_lds) && \
    __has_builtin(__builtin_amdgcn_s_wait_tensorcnt)
#define USE_TDM 1
#else
#define USE_TDM 0
#endif

#if USE_TDM
typedef unsigned int u32x4 __attribute__((ext_vector_type(4)));
typedef int          i32x8 __attribute__((ext_vector_type(8)));
typedef int          i32x4 __attribute__((ext_vector_type(4)));

__device__ __forceinline__ unsigned lds_addr_of(const void* p) {
  // generic -> LDS(addrspace 3) cast yields the DS-space byte address
  return (unsigned)(unsigned long long)(__attribute__((address_space(3))) const char*)p;
}

// TDM: contiguous copy of ndw dwords (ndw <= 65535) global -> LDS.
// D# per CDNA5 ISA ch.8: 1-row 2D tile, data_size = 4 bytes.
__device__ __forceinline__ void tdm_load_dwords(const void* gsrc, void* ldst,
                                                unsigned ndw) {
  unsigned long long ga = (unsigned long long)gsrc;
  u32x4 g0;
  g0[0] = 1u;                                                // count=1 (valid)
  g0[1] = lds_addr_of(ldst);                                 // lds_addr
  g0[2] = (unsigned)ga;                                      // global[31:0]
  g0[3] = (unsigned)((ga >> 32) & 0x01ffffffu) | (2u << 30); // [56:32]|type=2
  i32x8 g1;
  g1[0] = (int)(2u << 16);                    // wg_mask=0, data_size=4B
  g1[1] = (int)((ndw & 0xffffu) << 16);       // tensor_dim0[15:0]
  g1[2] = (int)(((ndw >> 16) & 0xffffu) | (1u << 16)); // dim0[31:16]|dim1=1
  g1[3] = (int)((ndw & 0xffffu) << 16);       // tile_dim0 = ndw
  g1[4] = 1;                                  // tile_dim1 = 1
  g1[5] = (int)ndw;                           // tensor_dim0_stride
  g1[6] = 0;
  g1[7] = 0;
  i32x4 z4 = {0, 0, 0, 0};
  i32x8 z8 = {0, 0, 0, 0, 0, 0, 0, 0};
  // 6-arg form (clang-23 / therock headers): groups 0..3 + extra group + cpol
  __builtin_amdgcn_tensor_load_to_lds(g0, g1, z4, z4, z8, 0);
}
#endif

// Issue stage (wave 0 only; TDM ignores EXEC, TENSORcnt is per-wave).
__device__ __forceinline__ void stage_to_lds(const void* g, void* s,
                                             unsigned ndw, int tid, int nthr) {
#if USE_TDM
  if (tid < 32) tdm_load_dwords(g, s, ndw);
#else
  for (unsigned i = (unsigned)tid; i < ndw; i += (unsigned)nthr)
    ((int*)s)[i] = ((const int*)g)[i];
#endif
}
__device__ __forceinline__ void stage_wait(int tid) {
#if USE_TDM
  if (tid < 32) __builtin_amdgcn_s_wait_tensorcnt(0);
#endif
  __syncthreads();
}

__device__ __forceinline__ v8f wmma_bf16(v16bf a, v16bf b, v8f c) {
  return __builtin_amdgcn_wmma_f32_16x16x32_bf16(false, a, false, b, (short)0, c,
                                                 false, false);
}

// A-matrix 16x32 bf16 fragment: lane L holds row (L&15); elems 0..7 -> K =
// kb + 8*hi + j ; elems 8..15 -> K = kb + 16 + 8*hi + j  (hi = L>=16).
__device__ __forceinline__ v16bf load_afrag(const float* row, int kb, int hi,
                                            float scale) {
  v16bf f;
  const int b0 = kb + 8 * hi, b1 = kb + 16 + 8 * hi;
#pragma unroll
  for (int j = 0; j < 8; j++) f[j] = (__bf16)(row[b0 + j] * scale);
#pragma unroll
  for (int j = 0; j < 8; j++) f[8 + j] = (__bf16)(row[b1 + j] * scale);
  return f;
}

__device__ __forceinline__ v16bf load_afrag2(const float* r0, const float* r1,
                                             int kb, int hi, float scale) {
  v16bf f;
  const int b0 = kb + 8 * hi, b1 = kb + 16 + 8 * hi;
#pragma unroll
  for (int j = 0; j < 8; j++) f[j] = (__bf16)(r0[b0 + j] * r1[b0 + j] * scale);
#pragma unroll
  for (int j = 0; j < 8; j++) f[8 + j] = (__bf16)(r0[b1 + j] * r1[b1 + j] * scale);
  return f;
}

// B-matrix 32x16 bf16 fragment from LDS panel: lane = column (nt*16 + L&15),
// elems 0..7 -> K = kt*32 + 8*hi + j ; elems 8..15 -> K = kt*32 + 16 + 8*hi + j.
__device__ __forceinline__ v16bf load_bfrag(const __bf16* W, int ld, int kt,
                                            int nt, int lane) {
  const __bf16* p = W + (size_t)(kt * 32 + 8 * (lane >> 4)) * ld + nt * 16 + (lane & 15);
  v16bf f;
#pragma unroll
  for (int j = 0; j < 8; j++) f[j] = p[(size_t)j * ld];
#pragma unroll
  for (int j = 0; j < 8; j++) f[8 + j] = p[(size_t)(16 + j) * ld];
  return f;
}

// ordered int/uint trick for float atomic max (init to -inf)
__device__ __forceinline__ void atomicMaxFloat(float* a, float v) {
  if (v >= 0.f)
    atomicMax((int*)a, __float_as_int(v));
  else
    atomicMin((unsigned int*)a, (unsigned int)__float_as_int(v));
}

__device__ __forceinline__ float sigmoidf_(float x) {
  return 1.f / (1.f + __expf(-x));
}

// ---------------------------------------------------------------- prep ----
__global__ void prep_weights(const float* Wq, const float* Wk, const float* Wv,
                             const float* W1, __bf16* Wqb, __bf16* Wkb,
                             __bf16* Wvb, __bf16* W1b) {
  int i = blockIdx.x * blockDim.x + threadIdx.x;
  if (i < HID * HID) {
    Wqb[i] = (__bf16)Wq[i];
    Wkb[i] = (__bf16)Wk[i];
    Wvb[i] = (__bf16)Wv[i];
  }
  if (i < 257 * HID) W1b[i] = (__bf16)W1[i];
}

__global__ void init_bufs(float* segmax, float* denom, float* out) {
  int i = blockIdx.x * blockDim.x + threadIdx.x;
  if (i < NN * NH) {
    segmax[i] = -__int_as_float(0x7f800000);  // -inf
    denom[i] = 0.f;
  }
  if (i < NN * HID) out[i] = 0.f;
}

// ------------------------------------------------------- Q = x*Wq + bq ----
__global__ void __launch_bounds__(128) qproj_kernel(const float* __restrict__ x,
                                                    const __bf16* __restrict__ Wqb,
                                                    const float* __restrict__ bq,
                                                    float* __restrict__ Q) {
  __shared__ __align__(16) __bf16 sWq[HID * HID];  // 32 KB
  const int tid = threadIdx.x;
  const int lane = tid & 31;
  const int wave = tid >> 5;
  const int l15 = lane & 15, hi = lane >> 4;
  const int tile = blockIdx.x * 4 + wave;  // 16 nodes per tile
  const int row0 = tile * 16;

  stage_to_lds(Wqb, sWq, (HID * HID * 2) / 4, tid, 128);

  int node = row0 + l15;
  int nclamp = node < NN ? node : NN - 1;
  const float* xr = x + (size_t)nclamp * HID;
  v16bf a[4];
#pragma unroll
  for (int t = 0; t < 4; t++) a[t] = load_afrag(xr, t * 32, hi, 1.f);

  stage_wait(tid);

#pragma unroll
  for (int nt = 0; nt < 8; ++nt) {
    v8f acc = {};
#pragma unroll
    for (int t = 0; t < 4; t++) {
      v16bf b = load_bfrag(sWq, HID, t, nt, lane);
      acc = wmma_bf16(a[t], b, acc);
    }
    const float bias = bq[nt * 16 + l15];  // column bias
#pragma unroll
    for (int i = 0; i < 8; i++) {
      int rnode = row0 + i + 8 * hi;  // D: vgpr i, lane half -> row
      if (rnode < NN) Q[(size_t)rnode * HID + nt * 16 + l15] = acc[i] + bias;
    }
  }
}

// ------------------------------------------------- per-edge WMMA passes ----
// PASS 0: scores + atomicMax segmax.  PASS 1: e=exp(s-max), V accum + denom.
template <int PASS>
__global__ void __launch_bounds__(128) edge_kernel(
    const float* __restrict__ x, const float* __restrict__ ts,
    const int* __restrict__ src, const int* __restrict__ dst,
    const int* __restrict__ etyp, const float* __restrict__ eti,
    const float* __restrict__ rel, const __bf16* __restrict__ W1b,
    const float* __restrict__ b1, const float* __restrict__ W2,
    const float* __restrict__ b2, const float* __restrict__ W1f,
    const __bf16* __restrict__ Wpb, const float* __restrict__ bp,
    const float* __restrict__ tc, const float* __restrict__ Q,
    float* __restrict__ scores, float* __restrict__ segmax,
    float* __restrict__ denom, float* __restrict__ out) {
  __shared__ __align__(16) __bf16 sW1[257 * HID];  // 65,792 B
  __shared__ __align__(16) __bf16 sWp[HID * HID];  // 32,768 B
  __shared__ float lds_dw[4][16];
  __shared__ float lds_tm[4][16];
  __shared__ int lds_dst[4][16];
  __shared__ __align__(16) float lds_q[4][16][HID];  // PASS 0 (32 KB)
  __shared__ float lds_e[4][16][NH];                 // PASS 1

  const int tid = threadIdx.x;
  const int lane = tid & 31;
  const int wave = tid >> 5;
  const int l15 = lane & 15, hi = lane >> 4;
  const long tileb = ((long)blockIdx.x * 4 + wave) * 16;

  // TDM-stage the shared B-panels while we do the per-edge gathers.
  stage_to_lds(W1b, sW1, (257 * HID * 2) / 4, tid, 128);
  stage_to_lds(Wpb, sWp, (HID * HID * 2) / 4, tid, 128);

  long e = tileb + l15;
  int ec = (e < NE) ? (int)e : NE - 1;
  const int si = src[ec], di = dst[ec], rt = etyp[ec];
  const float tcv = fabsf(tc[0]) + 1e-9f;
  const float tm = sigmoidf_((ts[di] - eti[ec]) / tcv);
  if (hi == 0) {
    lds_tm[wave][l15] = tm;
    lds_dst[wave][l15] = di;
  }
  const float* rs = x + (size_t)si * HID;
  const float* rr = rel + (size_t)rt * HID;
  v16bf aS[4], aR[4], aP[4];
#pragma unroll
  for (int t = 0; t < 4; t++) {
    aS[t] = load_afrag(rs, t * 32, hi, 1.f);      // src_emb
    aR[t] = load_afrag(rr, t * 32, hi, tm);       // rel_emb = rel*tm
    aP[t] = load_afrag2(rs, rr, t * 32, hi, tm);  // src*rel*tm
  }

  stage_wait(tid);  // TENSORcnt drain (wave 0) + block barrier

  // h = relu([src,rel,tm] @ W1 + b1); part[i] accumulates h . W2 per row
  float part[8];
#pragma unroll
  for (int i = 0; i < 8; i++) part[i] = 0.f;
  const float b2v = b2[0];
#pragma unroll
  for (int nt = 0; nt < 8; ++nt) {
    v8f acc = {};
#pragma unroll
    for (int t = 0; t < 8; t++) {
      v16bf b = load_bfrag(sW1, HID, t, nt, lane);
      acc = wmma_bf16(t < 4 ? aS[t] : aR[t - 4], b, acc);
    }
    const float bias = b1[nt * 16 + l15];
    const float w1last = W1f[256 * HID + nt * 16 + l15];  // K=256 (time_mask)
    const float w2c = W2[nt * 16 + l15];
#pragma unroll
    for (int i = 0; i < 8; i++) {
      float h = acc[i] + bias + lds_tm[wave][i + 8 * hi] * w1last;
      h = fmaxf(h, 0.f);
      part[i] += h * w2c;
    }
  }
#pragma unroll
  for (int i = 0; i < 8; i++) {
    float s = part[i];
    s += __shfl_xor(s, 1);
    s += __shfl_xor(s, 2);
    s += __shfl_xor(s, 4);
    s += __shfl_xor(s, 8);
    if (l15 == 0) lds_dw[wave][i + 8 * hi] = sigmoidf_(s + b2v);
  }
  __syncthreads();

  // k (PASS 0) or v (PASS 1): dw[m] * (prod @ Wp)[m][n] + bp[n]
  float dwr[8];
#pragma unroll
  for (int i = 0; i < 8; i++) dwr[i] = lds_dw[wave][i + 8 * hi];
  v8f pout[8];
#pragma unroll
  for (int nt = 0; nt < 8; ++nt) {
    v8f acc = {};
#pragma unroll
    for (int t = 0; t < 4; t++) {
      v16bf b = load_bfrag(sWp, HID, t, nt, lane);
      acc = wmma_bf16(aP[t], b, acc);
    }
    const float bias = bp[nt * 16 + l15];
#pragma unroll
    for (int i = 0; i < 8; i++) acc[i] = acc[i] * dwr[i] + bias;
    pout[nt] = acc;
  }

  if (PASS == 0) {
    // stage q rows for the 16 edges of this wave tile
#pragma unroll 4
    for (int row = 0; row < 16; ++row) {
      long ee = tileb + row;
      int eec = (ee < NE) ? (int)ee : NE - 1;
      int dd = dst[eec];
      ((float4*)lds_q[wave][row])[lane] =
          ((const float4*)(Q + (size_t)dd * HID))[lane];
    }
    __syncthreads();
    // score[m][head] = (q[m] . k[m])_head / sqrt(16)
#pragma unroll
    for (int nt = 0; nt < 8; ++nt) {
#pragma unroll
      for (int i = 0; i < 8; i++) {
        const int m = i + 8 * hi;
        float p = pout[nt][i] * lds_q[wave][m][nt * 16 + l15];
        p += __shfl_xor(p, 1);
        p += __shfl_xor(p, 2);
        p += __shfl_xor(p, 4);
        p += __shfl_xor(p, 8);
        if (l15 == 0) {
          long em = tileb + m;
          if (em < NE) {
            float s = p * 0.25f;
            scores[em * NH + nt] = s;
            atomicMaxFloat(&segmax[(size_t)lds_dst[wave][m] * NH + nt], s);
          }
        }
      }
    }
  } else {
    // e = exp(score - segmax); accumulate denom and v*e
    if (hi == 0) {
      long em = tileb + l15;
      bool v2 = em < NE;
      long emc = v2 ? em : NE - 1;
      int dd = lds_dst[wave][l15];
#pragma unroll
      for (int nt = 0; nt < 8; nt++) {
        float ev = __expf(scores[emc * NH + nt] - segmax[(size_t)dd * NH + nt]);
        lds_e[wave][l15][nt] = ev;
        if (v2) atomicAdd(&denom[(size_t)dd * NH + nt], ev);
      }
    }
    __syncthreads();
#pragma unroll
    for (int nt = 0; nt < 8; ++nt) {
#pragma unroll
      for (int i = 0; i < 8; i++) {
        const int m = i + 8 * hi;
        long em = tileb + m;
        if (em < NE) {
          float val = pout[nt][i] * lds_e[wave][m][nt];
          atomicAdd(&out[(size_t)lds_dst[wave][m] * HID + nt * 16 + l15], val);
        }
      }
    }
  }
}

__global__ void finalize(float* __restrict__ out, const float* __restrict__ denom) {
  int i = blockIdx.x * blockDim.x + threadIdx.x;
  if (i < NN * HID) {
    int node = i >> 7;
    int head = (i >> 4) & 7;
    float d = denom[node * NH + head];
    out[i] = (d > 0.f) ? out[i] / d : 0.f;
  }
}

// ------------------------------------------------------------- launch ----
extern "C" void kernel_launch(void* const* d_in, const int* in_sizes, int n_in,
                              void* d_out, int out_size, void* d_ws,
                              size_t ws_size, hipStream_t stream) {
  const float* x = (const float*)d_in[0];
  const float* ts = (const float*)d_in[1];
  const int* src = (const int*)d_in[2];
  const int* dst = (const int*)d_in[3];
  const int* etyp = (const int*)d_in[4];
  const float* eti = (const float*)d_in[5];
  const float* rel = (const float*)d_in[6];
  const float* Wq = (const float*)d_in[7];
  const float* bq = (const float*)d_in[8];
  const float* Wk = (const float*)d_in[9];
  const float* bk = (const float*)d_in[10];
  const float* Wv = (const float*)d_in[11];
  const float* bv = (const float*)d_in[12];
  const float* W1 = (const float*)d_in[13];
  const float* b1 = (const float*)d_in[14];
  const float* W2 = (const float*)d_in[15];
  const float* b2 = (const float*)d_in[16];
  const float* tc = (const float*)d_in[17];
  float* out = (float*)d_out;

  char* ws = (char*)d_ws;
  float* Q = (float*)(ws + 0);              // 25,600,000 B
  float* scores = (float*)(ws + 25600000);  // 25,600,000 B
  float* segmax = (float*)(ws + 51200000);  //  1,600,000 B
  float* denom = (float*)(ws + 52800000);   //  1,600,000 B
  __bf16* Wqb = (__bf16*)(ws + 54400000);
  __bf16* Wkb = (__bf16*)(ws + 54432768);
  __bf16* Wvb = (__bf16*)(ws + 54465536);
  __bf16* W1b = (__bf16*)(ws + 54498304);   // ends ~54.57 MB

  prep_weights<<<(257 * HID + 255) / 256, 256, 0, stream>>>(Wq, Wk, Wv, W1, Wqb,
                                                            Wkb, Wvb, W1b);
  init_bufs<<<(NN * HID + 255) / 256, 256, 0, stream>>>(segmax, denom, out);
  qproj_kernel<<<(3125 + 3) / 4, 128, 0, stream>>>(x, Wqb, bq, Q);
  edge_kernel<0><<<NE / 64, 128, 0, stream>>>(x, ts, src, dst, etyp, eti, rel,
                                              W1b, b1, W2, b2, W1, Wkb, bk, tc,
                                              Q, scores, segmax, denom, out);
  edge_kernel<1><<<NE / 64, 128, 0, stream>>>(x, ts, src, dst, etyp, eti, rel,
                                              W1b, b1, W2, b2, W1, Wvb, bv, tc,
                                              Q, scores, segmax, denom, out);
  finalize<<<(NN * HID + 255) / 256, 256, 0, stream>>>(out, denom);
}
